// SA_Layer_19550691131522
// MI455X (gfx1250) — matrix operand order repaired and checked by reading, not compiled, not run on hardware
//
#include <hip/hip_runtime.h>
#include <hip/hip_bf16.h>

#define B_  2
#define C_  256
#define N_  4096
#define CG_ 64
#define F_  1024   // 4*C

typedef __attribute__((ext_vector_type(16))) _Float16 v16h;
typedef __attribute__((ext_vector_type(8)))  float    v8f;
typedef __attribute__((ext_vector_type(4)))  unsigned int u32x4;
typedef __attribute__((ext_vector_type(8)))  int i32x8;
typedef __attribute__((ext_vector_type(4)))  int i32x4;

union Op16 { v16h v; _Float16 h[16]; };

__device__ __forceinline__ v8f wmma_f16(const Op16& a, const Op16& b, v8f c) {
  // D = A(16x32,f16) * B(32x16,f16) + C(16x16,f32)
  return __builtin_amdgcn_wmma_f32_16x16x32_f16(false, a.v, false, b.v,
                                                (short)0, c, false, false);
}

// 16-bit A-matrix (16x32) slot->K mapping (contiguous, matches B):
// lanes 0-15: K=0..15
// lanes 16-31: K=16..31
__device__ __forceinline__ int a_kmap(int j, int hl) {
  return 16 * hl + j;
}

__device__ __forceinline__ float wave_sum(float v) {
#pragma unroll
  for (int o = 16; o > 0; o >>= 1) v += __shfl_xor(v, o, 32);
  return v;
}
__device__ __forceinline__ float wave_max(float v) {
#pragma unroll
  for (int o = 16; o > 0; o >>= 1) v = fmaxf(v, __shfl_xor(v, o, 32));
  return v;
}

// LDS byte offset of a shared-memory object (generic-pointer low 32 bits map to
// the LDS aperture offset on gfx1250).
__device__ __forceinline__ unsigned lds_off(const void* p) {
  return (unsigned)(unsigned long long)p;
}

// ---- Tensor Data Mover: 2-D tile (f16 elements) Global -> LDS --------------
// Builds D# groups per ISA 8.3-8.5: count=1, type=2(image), data_size=2B,
// tile = tile0 x tile1, tensor dims for OOB, dim0 stride in elements.
__device__ __forceinline__ void tdm_issue_2d(unsigned lds_addr, const void* gp,
                                             unsigned tile0, unsigned tile1,
                                             unsigned tdim0, unsigned tdim1,
                                             unsigned long long stride0) {
  unsigned long long ga = (unsigned long long)gp;
  u32x4 g0;
  g0[0] = 1u;                                     // count=1, user mode
  g0[1] = lds_addr;                               // lds_addr [63:32]
  g0[2] = (unsigned)(ga & 0xFFFFFFFFu);           // global_addr [95:64]
  g0[3] = (unsigned)((ga >> 32) & 0x01FFFFFFu)    // global_addr [120:96]
          | (2u << 30);                           // type=2 [127:126]
  i32x8 g1;
  g1[0] = (int)(1u << 16);                        // data_size=1 (2 bytes)
  g1[1] = (int)((tdim0 & 0xFFFFu) << 16);         // tensor_dim0[15:0]
  g1[2] = (int)(((tdim0 >> 16) & 0xFFFFu) | ((tdim1 & 0xFFFFu) << 16));
  g1[3] = (int)(((tdim1 >> 16) & 0xFFFFu) | ((tile0 & 0xFFFFu) << 16));
  g1[4] = (int)(tile1 & 0xFFFFu);                 // tile_dim1 (tile_dim2=0)
  g1[5] = (int)(unsigned)(stride0 & 0xFFFFFFFFu); // tensor_dim0_stride[31:0]
  g1[6] = (int)(unsigned)((stride0 >> 32) & 0xFFFFu); // stride[47:32]
  g1[7] = 0;
  i32x4 gz = {0, 0, 0, 0};
#if __clang_major__ >= 23
  i32x8 gz8 = {0, 0, 0, 0, 0, 0, 0, 0};
  __builtin_amdgcn_tensor_load_to_lds(g0, g1, gz, gz, gz8, 0);
#else
  __builtin_amdgcn_tensor_load_to_lds(g0, g1, gz, gz, 0);
#endif
}
__device__ __forceinline__ void tdm_wait() {
  __builtin_amdgcn_s_wait_tensorcnt(0);
}
__device__ __forceinline__ void tdm_load_2d(unsigned lds_addr, const void* gp,
                                            unsigned tile0, unsigned tile1,
                                            unsigned tdim0, unsigned tdim1,
                                            unsigned long long stride0) {
  tdm_issue_2d(lds_addr, gp, tile0, tile1, tdim0, tdim1, stride0);
  tdm_wait();
}

// ---------------- kernel 0: convert MLP weights to f16 ----------------
__global__ __launch_bounds__(256) void k_cvt(const float* __restrict__ w1,
                                             const float* __restrict__ w2,
                                             _Float16* __restrict__ w1h,
                                             _Float16* __restrict__ w2h) {
  int i = blockIdx.x * 256 + threadIdx.x;
  if (i < C_ * F_) {
    w1h[i] = (_Float16)w1[i];
    w2h[i] = (_Float16)w2[i];
  }
}

// ---------------- kernel 1: qk (both layouts) + xv, f16 ----------------
__global__ __launch_bounds__(256) void k_qkv(const float* __restrict__ x,
                                             const float* __restrict__ Wqk,
                                             const float* __restrict__ Wv,
                                             _Float16* __restrict__ qk_cn,
                                             _Float16* __restrict__ qk_nc,
                                             _Float16* __restrict__ xvh) {
  const int n = blockIdx.x * 256 + threadIdx.x;
  const int c = blockIdx.y;
  const int b = blockIdx.z;
  const int g = c >> 6, o = c & 63;
  const float* xb = x + (size_t)b * C_ * N_;
  const float* wq = Wqk + ((size_t)g * CG_ + o) * CG_;
  float aq = 0.f;
#pragma unroll 8
  for (int i = 0; i < CG_; ++i) aq += wq[i] * xb[(size_t)(g * CG_ + i) * N_ + n];
  const float* wv = Wv + (size_t)c * C_;
  float av = 0.f;
#pragma unroll 8
  for (int cc = 0; cc < C_; ++cc) av += wv[cc] * xb[(size_t)cc * N_ + n];
  qk_cn[((size_t)b * C_ + c) * N_ + n] = (_Float16)aq;
  qk_nc[((size_t)b * N_ + n) * C_ + c] = (_Float16)aq;
  xvh  [((size_t)b * C_ + c) * N_ + n] = (_Float16)av;
}

// ---------------- kernel 2: energy = qk^T qk (TDM + WMMA) ----------------
// grid (N/64, N/64, B), block 128 (4 waves); TDM stages full 256x64 B panel.
__global__ __launch_bounds__(128) void k_energy(const _Float16* __restrict__ qk_nc,
                                                const _Float16* __restrict__ qk_cn,
                                                float* __restrict__ E) {
  __shared__ _Float16 lb[C_ * 64];
  const int wave = threadIdx.x >> 5, lane = threadIdx.x & 31;
  const int lm = lane & 15, hl = lane >> 4;
  const int b = blockIdx.z;
  const int n0 = blockIdx.x * 64 + wave * 16;  // E row (GEMM M)
  const int m0 = blockIdx.y * 64;              // E col (GEMM N)

  if (wave == 0)
    tdm_load_2d(lds_off(lb), qk_cn + (size_t)b * C_ * N_ + m0,
                /*tile*/64, C_, /*tensor*/N_, C_, /*stride0*/N_);
  __syncthreads();

  v8f acc[4] = {};
  const _Float16* Arow = qk_nc + ((size_t)b * N_ + n0 + lm) * C_;
  for (int c0 = 0; c0 < C_; c0 += 32) {
    Op16 a;
#pragma unroll
    for (int j = 0; j < 16; ++j) a.h[j] = Arow[c0 + a_kmap(j, hl)];
#pragma unroll
    for (int t = 0; t < 4; ++t) {
      Op16 bt;
#pragma unroll
      for (int j = 0; j < 16; ++j)
        bt.h[j] = lb[(c0 + j + 16 * hl) * 64 + t * 16 + lm];
      acc[t] = wmma_f16(a, bt, acc[t]);
    }
  }
#pragma unroll
  for (int t = 0; t < 4; ++t)
#pragma unroll
    for (int r = 0; r < 8; ++r)
      E[((size_t)b * N_ + n0 + r + 8 * hl) * N_ + m0 + t * 16 + lm] = acc[t][r];
}

// ---------------- kernel 3: row softmax, write f16 attn in place ----------------
__global__ __launch_bounds__(256) void k_softmax(float* __restrict__ E) {
  __shared__ float red[8];
  const int tid = threadIdx.x, wave = tid >> 5, lane = tid & 31;
  float* row = E + (size_t)blockIdx.x * N_;
  float vals[16];
  float mx = -3.0e38f;
#pragma unroll
  for (int i = 0; i < 16; ++i) { vals[i] = row[tid + 256 * i]; mx = fmaxf(mx, vals[i]); }
  mx = wave_max(mx);
  if (lane == 0) red[wave] = mx;
  __syncthreads();
#pragma unroll
  for (int w = 0; w < 8; ++w) mx = fmaxf(mx, red[w]);
  float s = 0.f;
#pragma unroll
  for (int i = 0; i < 16; ++i) { vals[i] = __expf(vals[i] - mx); s += vals[i]; }
  s = wave_sum(s);
  __syncthreads();
  if (lane == 0) red[wave] = s;
  __syncthreads();
  float tot = 0.f;
#pragma unroll
  for (int w = 0; w < 8; ++w) tot += red[w];
  const float inv = 1.f / tot;
  _Float16* arow = (_Float16*)row;  // alias: f16 row fits in first half of own f32 row
#pragma unroll
  for (int i = 0; i < 16; ++i) arow[tid + 256 * i] = (_Float16)(vals[i] * inv);
}

// ---- kernel 4: x_r = xv * attn (double-buffered TDM + WMMA), + residual ----
// grid (N/64, C/64, B), block 128; TDM prefetches 256-row chunk i+1 while
// WMMA consumes chunk i from the other LDS buffer.
__global__ __launch_bounds__(128) void k_attnv(const _Float16* __restrict__ xvh,
                                               const float* __restrict__ E,
                                               const float* __restrict__ x,
                                               float* __restrict__ yattn) {
  __shared__ _Float16 lb[2][256 * 64];  // 64 KiB
  const int wave = threadIdx.x >> 5, lane = threadIdx.x & 31;
  const int lm = lane & 15, hl = lane >> 4;
  const int b = blockIdx.z;
  const int c0 = blockIdx.y * 64 + wave * 16;  // GEMM M = channel
  const int m0 = blockIdx.x * 64;              // GEMM N = output position
  const _Float16* att = (const _Float16*)E;    // f16 attn aliased over f32 rows
  const size_t attbase = (size_t)b * N_ * (size_t)(2 * N_);
  v8f acc[4] = {};
  const _Float16* Arow = xvh + ((size_t)b * C_ + c0 + lm) * N_;

  const int NCH = N_ / 256;  // 16 chunks
  if (wave == 0) {
    tdm_issue_2d(lds_off(lb[0]), att + attbase + m0,
                 /*tile*/64, 256, /*tensor*/2 * N_, N_, /*stride0*/2 * N_);
    tdm_wait();
  }
  __syncthreads();
  for (int ch = 0; ch < NCH; ++ch) {
    if (wave == 0 && ch + 1 < NCH)  // prefetch next chunk into alternate buffer
      tdm_issue_2d(lds_off(lb[(ch + 1) & 1]),
                   att + attbase + (size_t)(ch + 1) * 256 * (2 * N_) + m0,
                   64, 256, 2 * N_, N_, 2 * N_);
    const _Float16* lbc = lb[ch & 1];
    for (int kk = 0; kk < 256; kk += 32) {
      const int n0g = ch * 256 + kk;
      __builtin_prefetch(Arow + n0g + 128, 0, 1);
      Op16 a;
#pragma unroll
      for (int j = 0; j < 16; ++j) a.h[j] = Arow[n0g + a_kmap(j, hl)];
#pragma unroll
      for (int t = 0; t < 4; ++t) {
        Op16 bt;
#pragma unroll
        for (int j = 0; j < 16; ++j)
          bt.h[j] = lbc[(kk + j + 16 * hl) * 64 + t * 16 + lm];
        acc[t] = wmma_f16(a, bt, acc[t]);
      }
    }
    if (ch + 1 < NCH) {
      if (wave == 0) tdm_wait();  // chunk i+1 DMA complete
      __syncthreads();
    }
  }
#pragma unroll
  for (int t = 0; t < 4; ++t)
#pragma unroll
    for (int r = 0; r < 8; ++r) {
      const int c = c0 + r + 8 * hl, m = m0 + t * 16 + lm;
      yattn[((size_t)b * N_ + m) * C_ + c] =
          x[((size_t)b * C_ + c) * N_ + m] + acc[t][r] * 0.0625f;
    }
}

// ---------------- kernel 5: LayerNorm 1 (wave per position) ----------------
__global__ __launch_bounds__(256) void k_ln1(const float* __restrict__ yattn,
                                             const float* __restrict__ g,
                                             const float* __restrict__ bb,
                                             float* __restrict__ yln_f,
                                             _Float16* __restrict__ yln_h) {
  const int wave = threadIdx.x >> 5, lane = threadIdx.x & 31;
  const size_t pos = (size_t)blockIdx.x * 8 + wave;
  const float* row = yattn + pos * C_;
  float v[8], s = 0.f;
#pragma unroll
  for (int i = 0; i < 8; ++i) { v[i] = row[lane + 32 * i]; s += v[i]; }
  const float mean = wave_sum(s) * (1.f / C_);
  float var = 0.f;
#pragma unroll
  for (int i = 0; i < 8; ++i) { float d = v[i] - mean; var += d * d; }
  var = wave_sum(var) * (1.f / C_);
  const float rin = rsqrtf(var + 1e-6f);
#pragma unroll
  for (int i = 0; i < 8; ++i) {
    const int c = lane + 32 * i;
    const float y = (v[i] - mean) * rin * g[c] + bb[c];
    yln_f[pos * C_ + c] = y;
    yln_h[pos * C_ + c] = (_Float16)y;
  }
}

// ---------------- kernel 6: fc1 + ReLU (TDM + WMMA) ----------------
// grid (F/64, B*N/64), block 128; TDM stages full 256x64 weight panel.
__global__ __launch_bounds__(128) void k_fc1(const _Float16* __restrict__ yln_h,
                                             const _Float16* __restrict__ w1h,
                                             const float* __restrict__ fc1_b,
                                             _Float16* __restrict__ h_h) {
  __shared__ _Float16 lb[C_ * 64];
  const int wave = threadIdx.x >> 5, lane = threadIdx.x & 31;
  const int lm = lane & 15, hl = lane >> 4;
  const int r0 = blockIdx.y * 64 + wave * 16;
  const int f0 = blockIdx.x * 64;

  if (wave == 0)
    tdm_load_2d(lds_off(lb), w1h + f0, /*tile*/64, C_, /*tensor*/F_, C_, /*stride0*/F_);
  __syncthreads();

  v8f acc[4] = {};
  const _Float16* Arow = yln_h + (size_t)(r0 + lm) * C_;
  for (int c0 = 0; c0 < C_; c0 += 32) {
    Op16 a;
#pragma unroll
    for (int j = 0; j < 16; ++j) a.h[j] = Arow[c0 + a_kmap(j, hl)];
#pragma unroll
    for (int t = 0; t < 4; ++t) {
      Op16 bt;
#pragma unroll
      for (int j = 0; j < 16; ++j)
        bt.h[j] = lb[(c0 + j + 16 * hl) * 64 + t * 16 + lm];
      acc[t] = wmma_f16(a, bt, acc[t]);
    }
  }
#pragma unroll
  for (int t = 0; t < 4; ++t)
#pragma unroll
    for (int r = 0; r < 8; ++r) {
      const int p = r0 + r + 8 * hl, f = f0 + t * 16 + lm;
      h_h[(size_t)p * F_ + f] = (_Float16)fmaxf(acc[t][r] + fc1_b[f], 0.f);
    }
}

// ---- kernel 7: fc2 + bias + residual (double-buffered TDM + WMMA) ----
// grid (C/64, B*N/64), block 128; prefetch next 256-row weight chunk.
__global__ __launch_bounds__(128) void k_fc2(const _Float16* __restrict__ h_h,
                                             const _Float16* __restrict__ w2h,
                                             const float* __restrict__ fc2_b,
                                             const float* __restrict__ yln_f,
                                             float* __restrict__ z) {
  __shared__ _Float16 lb[2][256 * 64];  // 64 KiB
  const int wave = threadIdx.x >> 5, lane = threadIdx.x & 31;
  const int lm = lane & 15, hl = lane >> 4;
  const int r0 = blockIdx.y * 64 + wave * 16;
  const int c0 = blockIdx.x * 64;
  v8f acc[4] = {};
  const _Float16* Arow = h_h + (size_t)(r0 + lm) * F_;

  const int NCH = F_ / 256;  // 4 chunks
  if (wave == 0) {
    tdm_issue_2d(lds_off(lb[0]), w2h + c0, /*tile*/64, 256,
                 /*tensor*/C_, F_, /*stride0*/C_);
    tdm_wait();
  }
  __syncthreads();
  for (int ch = 0; ch < NCH; ++ch) {
    if (wave == 0 && ch + 1 < NCH)
      tdm_issue_2d(lds_off(lb[(ch + 1) & 1]),
                   w2h + (size_t)(ch + 1) * 256 * C_ + c0,
                   64, 256, C_, F_, C_);
    const _Float16* lbc = lb[ch & 1];
    for (int kk = 0; kk < 256; kk += 32) {
      const int f0 = ch * 256 + kk;
      __builtin_prefetch(Arow + f0 + 128, 0, 1);
      Op16 a;
#pragma unroll
      for (int j = 0; j < 16; ++j) a.h[j] = Arow[f0 + a_kmap(j, hl)];
#pragma unroll
      for (int t = 0; t < 4; ++t) {
        Op16 bt;
#pragma unroll
        for (int j = 0; j < 16; ++j)
          bt.h[j] = lbc[(kk + j + 16 * hl) * 64 + t * 16 + lm];
        acc[t] = wmma_f16(a, bt, acc[t]);
      }
    }
    if (ch + 1 < NCH) {
      if (wave == 0) tdm_wait();
      __syncthreads();
    }
  }
#pragma unroll
  for (int t = 0; t < 4; ++t)
#pragma unroll
    for (int r = 0; r < 8; ++r) {
      const int p = r0 + r + 8 * hl, c = c0 + t * 16 + lm;
      z[(size_t)p * C_ + c] = yln_f[(size_t)p * C_ + c] + acc[t][r] + fc2_b[c];
    }
}

// ---------------- kernel 8: LayerNorm 2 + ReLU + transpose -> out ----------------
__global__ __launch_bounds__(256) void k_ln2out(const float* __restrict__ z,
                                                const float* __restrict__ g,
                                                const float* __restrict__ bb,
                                                float* __restrict__ out) {
  const int wave = threadIdx.x >> 5, lane = threadIdx.x & 31;
  const size_t pos = (size_t)blockIdx.x * 8 + wave;
  const int b = (int)(pos / N_), n = (int)(pos % N_);
  const float* row = z + pos * C_;
  float v[8], s = 0.f;
#pragma unroll
  for (int i = 0; i < 8; ++i) { v[i] = row[lane + 32 * i]; s += v[i]; }
  const float mean = wave_sum(s) * (1.f / C_);
  float var = 0.f;
#pragma unroll
  for (int i = 0; i < 8; ++i) { float d = v[i] - mean; var += d * d; }
  var = wave_sum(var) * (1.f / C_);
  const float rin = rsqrtf(var + 1e-6f);
#pragma unroll
  for (int i = 0; i < 8; ++i) {
    const int c = lane + 32 * i;
    const float y = fmaxf((v[i] - mean) * rin * g[c] + bb[c], 0.f);
    out[((size_t)b * C_ + c) * N_ + n] = y;
  }
}

extern "C" void kernel_launch(void* const* d_in, const int* in_sizes, int n_in,
                              void* d_out, int out_size, void* d_ws, size_t ws_size,
                              hipStream_t stream) {
  (void)in_sizes; (void)n_in; (void)out_size; (void)ws_size;
  const float* x     = (const float*)d_in[0];
  const float* Wqk   = (const float*)d_in[1];
  const float* Wv    = (const float*)d_in[2];
  const float* ln1_g = (const float*)d_in[3];
  const float* ln1_b = (const float*)d_in[4];
  const float* fc1_w = (const float*)d_in[5];
  const float* fc1_b = (const float*)d_in[6];
  const float* fc2_w = (const float*)d_in[7];
  const float* fc2_b = (const float*)d_in[8];
  const float* ln2_g = (const float*)d_in[9];
  const float* ln2_b = (const float*)d_in[10];
  float* out = (float*)d_out;
  char* ws = (char*)d_ws;

  _Float16* qk_cn = (_Float16*)(ws + 0x0000000);  // 4 MiB
  _Float16* qk_nc = (_Float16*)(ws + 0x0400000);  // 4 MiB
  _Float16* xvh   = (_Float16*)(ws + 0x0800000);  // 4 MiB
  _Float16* w1h   = (_Float16*)(ws + 0x0C00000);  // 512 KiB
  _Float16* w2h   = (_Float16*)(ws + 0x0C80000);  // 512 KiB
  float*    yattn = (float*)   (ws + 0x0D00000);  // 8 MiB
  float*    yln_f = (float*)   (ws + 0x1500000);  // 8 MiB
  _Float16* yln_h = (_Float16*)(ws + 0x1D00000);  // 4 MiB
  _Float16* h_h   = (_Float16*)(ws + 0x2100000);  // 16 MiB
  float*    z     = (float*)   (ws + 0x3100000);  // 8 MiB
  float*    E     = (float*)   (ws + 0x3900000);  // 128 MiB

  k_cvt    <<<dim3((C_ * F_ + 255) / 256), 256, 0, stream>>>(fc1_w, fc2_w, w1h, w2h);
  k_qkv    <<<dim3(N_ / 256, C_, B_),      256, 0, stream>>>(x, Wqk, Wv, qk_cn, qk_nc, xvh);
  k_energy <<<dim3(N_ / 64, N_ / 64, B_),  128, 0, stream>>>(qk_nc, qk_cn, E);
  k_softmax<<<dim3(B_ * N_),               256, 0, stream>>>(E);
  k_attnv  <<<dim3(N_ / 64, C_ / 64, B_),  128, 0, stream>>>(xvh, E, x, yattn);
  k_ln1    <<<dim3((B_ * N_) / 8),         256, 0, stream>>>(yattn, ln1_g, ln1_b, yln_f, yln_h);
  k_fc1    <<<dim3(F_ / 64, (B_ * N_) / 64), 128, 0, stream>>>(yln_h, w1h, fc1_b, h_h);
  k_fc2    <<<dim3(C_ / 64, (B_ * N_) / 64), 128, 0, stream>>>(h_h, w2h, fc2_b, yln_f, z);
  k_ln2out <<<dim3((B_ * N_) / 8),         256, 0, stream>>>(z, ln2_g, ln2_b, out);
}
